// MANN_network_18348100288740
// MI455X (gfx1250) — matrix operand order repaired and checked by reading, not compiled, not run on hardware
//
#include <hip/hip_runtime.h>
#include <hip/hip_bf16.h>

// ---------------------------------------------------------------------------
// MoE-MANN forward for MI455X (gfx1250), wave32 + v_wmma_f32_16x16x32_bf16.
//
// B=8192, OBS=128, LAT=32, NXT=128, E=8, HG=256, H=256.
//
// Expert mixing sum_e coef[b,e] * (inp @ W_e) is done by scaling the A
// fragment rows with coef (v_pk_mul_bf16) and accumulating all experts into
// one f32 WMMA accumulator. Biases are folded in with an augmented 1-column.
// Weights are pre-swizzled into the WMMA B-fragment layout so every B
// fragment is two aligned 16B loads per lane; all strides / trip counts are
// template constants so B addresses fold into immediate offsets.
//
// MB row-bands are processed per block so each B fragment held in registers
// is reused for MB accumulator tiles: weight re-read traffic from L2 (the
// dominant roofline term, ~1.6 GB/forward at MB=1) is divided by MB, while
// the grid keeps >=1024 waves for latency hiding. MB=2 balances VGPRs
// (~200/lane -> ~4 waves/SIMD) against traffic.
// ---------------------------------------------------------------------------

#define BB   8192
#define OBS  128
#define LAT  32
#define NE   8

typedef __attribute__((ext_vector_type(16))) __bf16 v16bf;
typedef __attribute__((ext_vector_type(8)))  float  v8f;

struct Frag {
    union {
        v16bf    v;
        uint4    q[2];
        unsigned u[8];
    };
};

// packed bf16 multiply (VOP3P V_PK_MUL_BF16) — scale two bf16 lanes at once
static __device__ inline unsigned pkmul_bf16(unsigned a, unsigned s) {
    unsigned d;
    asm volatile("v_pk_mul_bf16 %0, %1, %2" : "=v"(d) : "v"(a), "v"(s));
    return d;
}

static __device__ inline __bf16 to_bf16(float f) { return (__bf16)f; }

// ---------------------------------------------------------------------------
// Activation prep: build augmented bf16 activation buffers.
//  act0 [B,192]: cols 0..127 = x, 128..159 = z, 160 = 1.0, 161..191 = 0
//  act1/act2 [B,320]: cols 256..287 = z, 288 = 1.0, 289..319 = 0 (rest by GEMMs)
//  g1/g2 [B,288]: col 256 = 1.0, 257..287 = 0 (rest by GEMMs)
// ---------------------------------------------------------------------------
__global__ void prep_act(const float* __restrict__ x, const float* __restrict__ z,
                         __bf16* __restrict__ act0, __bf16* __restrict__ act1,
                         __bf16* __restrict__ act2, __bf16* __restrict__ g1,
                         __bf16* __restrict__ g2) {
    const int b   = blockIdx.x;
    const int tid = threadIdx.x;   // blockDim.x == 64

    for (int c = tid; c < 192; c += 64) {
        float v = (c < OBS)       ? x[(size_t)b * OBS + c]
                : (c < OBS + LAT) ? z[(size_t)b * LAT + (c - OBS)]
                : (c == 160)      ? 1.0f : 0.0f;
        act0[(size_t)b * 192 + c] = to_bf16(v);
    }
    {   // act1 / act2 tails (cols 256..319)
        int col = 256 + tid;
        float v = (tid < LAT) ? z[(size_t)b * LAT + tid] : (col == 288 ? 1.0f : 0.0f);
        act1[(size_t)b * 320 + col] = to_bf16(v);
        act2[(size_t)b * 320 + col] = to_bf16(v);
    }
    if (tid < 32) {  // g tails (cols 256..287)
        int col = 256 + tid;
        float v = (col == 256) ? 1.0f : 0.0f;
        g1[(size_t)b * 288 + col] = to_bf16(v);
        g2[(size_t)b * 288 + col] = to_bf16(v);
    }
}

// ---------------------------------------------------------------------------
// Weight swizzle: f32 [E,K,N] row-major (+ bias [E,N] appended as row K,
// zero padding to Kpad=32*nkc, Npad=16*nnt) -> bf16 WMMA B-fragment layout.
// Tile (e,nt,kc) is 512 elems; within a tile: lane-major 16 contiguous bf16
// per lane. Lane (h=lane/16, nl=lane%16) element t holds B[k=32*kc+16h+t][n].
// ---------------------------------------------------------------------------
__global__ void swizzle_weights(const float* __restrict__ W, const float* __restrict__ bias,
                                __bf16* __restrict__ dst, int E, int K, int N,
                                int nkc, int nnt) {
    int total = E * nnt * nkc * 512;
    int idx = blockIdx.x * blockDim.x + threadIdx.x;
    if (idx >= total) return;
    int t    = idx & 15;
    int lane = (idx >> 4) & 31;
    int tile = idx >> 9;
    int kc = tile % nkc;
    int r  = tile / nkc;
    int nt = r % nnt;
    int e  = r / nnt;
    int hh = lane >> 4, nl = lane & 15;
    int k = kc * 32 + hh * 16 + t;
    int n = nt * 16 + nl;
    float v = 0.0f;
    if (n < N) {
        if (k < K)       v = W[((size_t)e * K + k) * N + n];
        else if (k == K) v = bias[(size_t)e * N + n];
    }
    dst[idx] = to_bf16(v);
}

// ---------------------------------------------------------------------------
// Fused MoE GEMM over MB 16-row bands:
//   out = act_elu( sum_e sum_k (coef_e * A) @ W_e )
// One wave owns TPW 16x16 output tiles (same columns) for each of MB bands;
// every B fragment in registers feeds MB WMMAs.
// A fragment layout: lane (m=lane%16,h=lane/16)
// holds K = [16h,16h+16) (matches swizzled B)  ->  two aligned b128 loads.
// E==1 -> plain GEMM (gating path, no coef scaling).
// ---------------------------------------------------------------------------
template <int MB, int TPW, int E, int NKC, int NNT, int LDA, int LDO,
          bool DO_ELU, bool OUT_BF16>
__global__ __launch_bounds__(128) void moe_gemm(
    const __bf16* __restrict__ A, const __bf16* __restrict__ Ws,
    const float* __restrict__ coef,
    __bf16* __restrict__ outb, float* __restrict__ outf) {

    const int lane = threadIdx.x & 31;
    const int wave = threadIdx.x >> 5;
    const int m  = lane & 15;
    const int hh = lane >> 4;
    const int row0 = blockIdx.x * (16 * MB);

    const __bf16* __restrict__ Arow  = A + (size_t)(row0 + m) * LDA + 16 * hh;
    // fold wave's tile offset + lane offset into one base; everything else is
    // a compile-time-constant immediate offset
    const __bf16* __restrict__ Wbase =
        Ws + (size_t)(wave * TPW) * NKC * 512 + lane * 16;

    float coefs[MB][(E > 1) ? E : 1];
    if (E > 1) {
#pragma unroll
        for (int b = 0; b < MB; ++b) {
            const float* cp = coef + (size_t)(row0 + b * 16 + m) * NE;
#pragma unroll
            for (int e = 0; e < E; ++e) coefs[b][e] = cp[e];
        }
    }

    auto loadA = [&](Frag* f, int kc) {
#pragma unroll
        for (int b = 0; b < MB; ++b) {
            const __bf16* ap = Arow + (size_t)b * 16 * LDA + kc * 32;
            f[b].q[0] = *(const uint4*)(const void*)(ap);
            f[b].q[1] = *(const uint4*)(const void*)(ap + 8);
        }
    };
    auto loadB = [&](Frag* f, int kc, int e) {
#pragma unroll
        for (int t = 0; t < TPW; ++t) {
            const __bf16* bp = Wbase + ((size_t)(e * NNT + t) * NKC + kc) * 512;
            f[t].q[0] = *(const uint4*)(const void*)(bp);
            f[t].q[1] = *(const uint4*)(const void*)(bp + 8);
        }
    };
    auto scaleA = [&](const Frag& in, int b, int e) {
        Frag a;
        if (E > 1) {
            union { float f; unsigned u; } cu;
            cu.f = coefs[b][e];
            unsigned sb = cu.u >> 16;        // f32 -> bf16 (truncate)
            unsigned sp = sb | (sb << 16);   // broadcast into a pair
#pragma unroll
            for (int i = 0; i < 8; ++i) a.u[i] = pkmul_bf16(in.u[i], sp);
        } else {
            a = in;
        }
        return a;
    };

    v8f acc[MB][TPW] = {};
    Frag araw[MB] = {}, anext[MB] = {};
    Frag bcur[TPW] = {}, bnext[TPW] = {};

    loadA(araw, 0);
    loadB(bcur, 0, 0);

    for (int kc = 0; kc < NKC; ++kc) {
        if (kc + 1 < NKC) loadA(anext, kc + 1);
#pragma unroll
        for (int e = 0; e < E; ++e) {
            const bool lastGroup = (e == E - 1);
            if (!(lastGroup && kc == NKC - 1))
                loadB(bnext, lastGroup ? (kc + 1) : kc, lastGroup ? 0 : (e + 1));
#pragma unroll
            for (int b = 0; b < MB; ++b) {
                Frag a = scaleA(araw[b], b, e);
#pragma unroll
                for (int t = 0; t < TPW; ++t)
                    acc[b][t] = __builtin_amdgcn_wmma_f32_16x16x32_bf16(
                        false, a.v, false, bcur[t].v, (short)0, acc[b][t],
                        false, false);
            }
            // fully-unrolled e-loop: these copies are pure SSA renames
#pragma unroll
            for (int t = 0; t < TPW; ++t) bcur[t] = bnext[t];
        }
#pragma unroll
        for (int b = 0; b < MB; ++b) araw[b] = anext[b];
    }

    // C layout: lane element j -> row = 8h + j, col = lane%16 (ISA 7.12.2)
#pragma unroll
    for (int b = 0; b < MB; ++b) {
#pragma unroll
        for (int t = 0; t < TPW; ++t) {
            int nt  = wave * TPW + t;
            int col = nt * 16 + m;
#pragma unroll
            for (int j = 0; j < 8; ++j) {
                float v = acc[b][t][j];
                if (DO_ELU) v = (v > 0.0f) ? v : (__expf(v) - 1.0f);
                size_t idx = (size_t)(row0 + b * 16 + hh * 8 + j) * LDO + col;
                if (OUT_BF16) outb[idx] = to_bf16(v);
                else          outf[idx] = v;
            }
        }
    }
}

// ---------------------------------------------------------------------------
// Softmax over E=8 gating logits per row (logits buffer has stride 16).
// ---------------------------------------------------------------------------
__global__ void softmax8(const float* __restrict__ logits, float* __restrict__ coef) {
    int b = blockIdx.x * blockDim.x + threadIdx.x;
    if (b >= BB) return;
    float l[NE];
    float mx = -1e30f;
#pragma unroll
    for (int e = 0; e < NE; ++e) { l[e] = logits[(size_t)b * 16 + e]; mx = fmaxf(mx, l[e]); }
    float s = 0.0f;
#pragma unroll
    for (int e = 0; e < NE; ++e) { l[e] = __expf(l[e] - mx); s += l[e]; }
    float inv = 1.0f / s;
#pragma unroll
    for (int e = 0; e < NE; ++e) coef[(size_t)b * NE + e] = l[e] * inv;
}

// ---------------------------------------------------------------------------
extern "C" void kernel_launch(void* const* d_in, const int* in_sizes, int n_in,
                              void* d_out, int out_size, void* d_ws, size_t ws_size,
                              hipStream_t stream) {
    const float* x   = (const float*)d_in[0];
    const float* z   = (const float*)d_in[1];
    const float* gw0 = (const float*)d_in[2];  const float* gb0 = (const float*)d_in[3];
    const float* gw1 = (const float*)d_in[4];  const float* gb1 = (const float*)d_in[5];
    const float* gw2 = (const float*)d_in[6];  const float* gb2 = (const float*)d_in[7];
    const float* gw3 = (const float*)d_in[8];  const float* gb3 = (const float*)d_in[9];
    const float* w0  = (const float*)d_in[10]; const float* b0  = (const float*)d_in[11];
    const float* w1  = (const float*)d_in[12]; const float* b1  = (const float*)d_in[13];
    const float* w2  = (const float*)d_in[14]; const float* b2  = (const float*)d_in[15];

    char*  ws  = (char*)d_ws;
    size_t off = 0;
    auto alloc = [&](size_t bytes) -> void* {
        void* p = ws + off;
        off = (off + bytes + 255) & ~(size_t)255;
        return p;
    };

    // activations (bf16, augmented)
    __bf16* act0 = (__bf16*)alloc((size_t)BB * 192 * 2);
    __bf16* act1 = (__bf16*)alloc((size_t)BB * 320 * 2);
    __bf16* act2 = (__bf16*)alloc((size_t)BB * 320 * 2);
    __bf16* g1   = (__bf16*)alloc((size_t)BB * 288 * 2);
    __bf16* g2   = (__bf16*)alloc((size_t)BB * 288 * 2);
    float*  logi = (float*) alloc((size_t)BB * 16 * 4);
    float*  coef = (float*) alloc((size_t)BB * NE * 4);
    // swizzled bf16 weights (bias row appended, zero-padded)
    __bf16* gw0s = (__bf16*)alloc((size_t)1 * 16 * 6  * 512 * 2);
    __bf16* gw1s = (__bf16*)alloc((size_t)1 * 16 * 9  * 512 * 2);
    __bf16* gw2s = (__bf16*)alloc((size_t)1 * 16 * 9  * 512 * 2);
    __bf16* gw3s = (__bf16*)alloc((size_t)1 * 1  * 9  * 512 * 2);
    __bf16* w0s  = (__bf16*)alloc((size_t)8 * 16 * 6  * 512 * 2);
    __bf16* w1s  = (__bf16*)alloc((size_t)8 * 16 * 10 * 512 * 2);
    __bf16* w2s  = (__bf16*)alloc((size_t)8 * 8  * 10 * 512 * 2);

    prep_act<<<BB, 64, 0, stream>>>(x, z, act0, act1, act2, g1, g2);

    auto swz = [&](const float* W, const float* bias, __bf16* dst,
                   int E, int K, int N, int nkc, int nnt) {
        int total = E * nnt * nkc * 512;
        swizzle_weights<<<(total + 255) / 256, 256, 0, stream>>>(W, bias, dst, E, K, N, nkc, nnt);
    };
    swz(gw0, gb0, gw0s, 1, 160, 256, 6, 16);
    swz(gw1, gb1, gw1s, 1, 256, 256, 9, 16);
    swz(gw2, gb2, gw2s, 1, 256, 256, 9, 16);
    swz(gw3, gb3, gw3s, 1, 256, 8,   9, 1);
    swz(w0,  b0,  w0s,  8, 160, 256, 6, 16);
    swz(w1,  b1,  w1s,  8, 288, 256, 10, 16);
    swz(w2,  b2,  w2s,  8, 288, 128, 10, 8);

    const int blocks = BB / 32;  // MB=2 -> 256 blocks of 32 rows

    // gating MLP (E=1, bias via augmented column), ELU, bf16 out
    moe_gemm<2, 4, 1, 6,  16, 192, 288, true,  true >
        <<<blocks, 128, 0, stream>>>(act0, gw0s, nullptr, g1, nullptr);
    moe_gemm<2, 4, 1, 9,  16, 288, 288, true,  true >
        <<<blocks, 128, 0, stream>>>(g1, gw1s, nullptr, g2, nullptr);
    moe_gemm<2, 4, 1, 9,  16, 288, 288, true,  true >
        <<<blocks, 128, 0, stream>>>(g2, gw2s, nullptr, g1, nullptr);
    // gating head -> f32 logits [B,16] (cols 8..15 are zero-padded)
    moe_gemm<2, 1, 1, 9,  1,  288, 16,  false, false>
        <<<blocks, 32, 0, stream>>>(g1, gw3s, nullptr, nullptr, logi);
    softmax8<<<(BB + 255) / 256, 256, 0, stream>>>(logi, coef);

    // expert layers: coef-scaled A rows, 8 experts accumulated in one C
    moe_gemm<2, 4, 8, 6,  16, 192, 320, true,  true >
        <<<blocks, 128, 0, stream>>>(act0, w0s, coef, act1, nullptr);
    moe_gemm<2, 4, 8, 10, 16, 320, 320, true,  true >
        <<<blocks, 128, 0, stream>>>(act1, w1s, coef, act2, nullptr);
    moe_gemm<2, 2, 8, 10, 8,  320, 128, false, false>
        <<<blocks, 128, 0, stream>>>(act2, w2s, coef, nullptr, (float*)d_out);
}
